// MessageLayer_4217657885289
// MI455X (gfx1250) — compile-verified, matrix operand order checked by reading, not run on hardware
//
#include <hip/hip_runtime.h>
#include <hip/hip_bf16.h>
#include <math.h>

// ---------------------------------------------------------------------------
// Types for WMMA fragments (gfx1250 wave32, v_wmma_f32_16x16x32_f16)
// ---------------------------------------------------------------------------
typedef _Float16 v16h __attribute__((ext_vector_type(16)));
typedef _Float16 h8   __attribute__((ext_vector_type(8)));
typedef _Float16 h4   __attribute__((ext_vector_type(4)));
typedef float    v8f  __attribute__((ext_vector_type(8)));

union AFrag { v16h v; h8 h[2]; };

// Load a 16x32 f16 fragment (A or B) from a row-major matrix with leading
// dimension `ld` (in halves). ISA layout (CDNA5 ISA):
//   lane L: row = L%16, K-group base = (L/16)*8
//   halves 0..7  -> K = kg+0..7        (VGPRs 0..3)
//   halves 8..15 -> K = kg+16..kg+23   (VGPRs 4..7)
__device__ __forceinline__ v16h frag_rm(const _Float16* p, int ld, int lane) {
  int r  = lane & 15;
  int kg = (lane >> 4) << 3;
  AFrag f;
  f.h[0] = *(const h8*)(p + r * ld + kg);
  f.h[1] = *(const h8*)(p + r * ld + kg + 16);
  return f.v;
}

__device__ __forceinline__ v8f wmma_f16(v16h a, v16h b, v8f c) {
  return __builtin_amdgcn_wmma_f32_16x16x32_f16(false, a, false, b,
                                                (short)0, c, false, false);
}

// Sign-aware float atomic max via integer atomics (monotone bit tricks).
__device__ __forceinline__ void atomicMaxF(float* addr, float val) {
  if (val >= 0.0f) atomicMax((int*)addr, __float_as_int(val));
  else             atomicMin((unsigned int*)addr, __float_as_uint(val));
}

#define FEA 64
#define HID 256       // 4*FEA
#define IN2 128       // 2*FEA
#define GHID 256
#define TM  128       // edges per workgroup in the fused MLP kernel

// ---------------------------------------------------------------------------
// Kernel 0: convert + transpose weights to f16 (N-major so B fragments are
// contiguous 16B loads).  WinT: 256x128, WoutT: 64x256, gW1T: 256x64.
// ---------------------------------------------------------------------------
__global__ void convert_w_kernel(const float* __restrict__ Win,
                                 const float* __restrict__ Wout,
                                 const float* __restrict__ gW1,
                                 _Float16* __restrict__ WinT,
                                 _Float16* __restrict__ WoutT,
                                 _Float16* __restrict__ gW1T) {
  int i = blockIdx.x * 256 + threadIdx.x;
  if (i < 256 * 128) { int n = i >> 7, k = i & 127; WinT[i]  = (_Float16)Win[k * 256 + n]; }
  if (i < 64 * 256)  { int n = i >> 8, k = i & 255; WoutT[i] = (_Float16)Wout[k * 64 + n]; }
  if (i < 256 * 64)  { int n = i >> 6, k = i & 63;  gW1T[i]  = (_Float16)gW1[k * 256 + n]; }
}

// ---------------------------------------------------------------------------
// Kernel 1: init per-atom accumulators.
// ---------------------------------------------------------------------------
__global__ void init_kernel(float* __restrict__ gmax, float* __restrict__ denom,
                            float* __restrict__ pooled, int N) {
  int i = blockIdx.x * blockDim.x + threadIdx.x;
  if (i < N) { gmax[i] = -1e30f; denom[i] = 0.0f; }
  if (i < N * FEA) pooled[i] = 0.0f;
}

// ---------------------------------------------------------------------------
// Kernel 2: fused edge MLP + gate scalar, all GEMMs on WMMA.
//   X(128x128,f16) -> H1=relu(X@Win+b)(128x256) -> FEA=H1@Wout+b (128x64)
//   -> G1=relu(FEA@gW1+b)(128x256) -> gate = G1@gW2+b (folded epilogue)
// Each wave owns a 16-row strip of the 128-edge tile.
// ---------------------------------------------------------------------------
__global__ __launch_bounds__(256)
void edge_mlp_kernel(const float* __restrict__ atom_fea,
                     const int*   __restrict__ selfIdx,
                     const int*   __restrict__ nbrIdx,
                     const _Float16* __restrict__ WinT,
                     const _Float16* __restrict__ WoutT,
                     const _Float16* __restrict__ gW1T,
                     const float* __restrict__ b_in,
                     const float* __restrict__ b_out,
                     const float* __restrict__ gb1,
                     const float* __restrict__ gW2,
                     const float* __restrict__ gb2,
                     _Float16* __restrict__ feaH,
                     float* __restrict__ gateOut,
                     float* __restrict__ gmax,
                     int M) {
  __shared__ __align__(16) _Float16 Xs[TM * IN2];     // 32 KB
  __shared__ __align__(16) _Float16 H1s[TM * HID];    // 64 KB
  __shared__ __align__(16) _Float16 Fs[TM * FEA];     // 16 KB
  __shared__ float bin_s[HID], gb1_s[GHID], gw2_s[GHID], bout_s[FEA];
  __shared__ int   sidx_s[TM];

  const int tid  = threadIdx.x;
  const int lane = tid & 31;
  const int wave = tid >> 5;
  const int base = blockIdx.x * TM;

  // stage biases / gate weights / self indices
  if (tid < HID) { bin_s[tid] = b_in[tid]; gb1_s[tid] = gb1[tid]; gw2_s[tid] = gW2[tid]; }
  if (tid < FEA) bout_s[tid] = b_out[tid];
  if (tid < TM) {
    int e = base + tid; if (e >= M) e = M - 1;
    sidx_s[tid] = selfIdx[e];
  }

  // gather concat(atom[self], atom[nbr]) -> Xs as f16; 2 threads per edge
  {
    const int eloc = tid >> 1, half = tid & 1;
    int e = base + eloc; if (e >= M) e = M - 1;
    const int src = half ? nbrIdx[e] : selfIdx[e];
    const float4* sp = (const float4*)(atom_fea + (size_t)src * FEA);
    _Float16* dst = &Xs[eloc * IN2 + half * FEA];
    #pragma unroll
    for (int i = 0; i < 16; ++i) {
      float4 v = sp[i];
      h4 o = { (_Float16)v.x, (_Float16)v.y, (_Float16)v.z, (_Float16)v.w };
      *(h4*)(dst + i * 4) = o;
    }
  }
  __syncthreads();

  // ---- stage 1: H1 = relu(X @ Win + b_in)   M=128 N=256 K=128 ----
  {
    const _Float16* Arow = &Xs[(wave * 16) * IN2];
    const int n0 = lane & 15, mb = (lane >> 4) * 8;
    for (int nt = 0; nt < 16; ++nt) {
      v8f c = {};
      #pragma unroll
      for (int ks = 0; ks < 4; ++ks) {
        v16h a = frag_rm(Arow + ks * 32, IN2, lane);
        v16h b = frag_rm(WinT + (nt * 16) * IN2 + ks * 32, IN2, lane);
        c = wmma_f16(a, b, c);
      }
      const int n = nt * 16 + n0;
      const float bias = bin_s[n];
      #pragma unroll
      for (int r = 0; r < 8; ++r) {
        float v = c[r] + bias;
        v = v > 0.0f ? v : 0.0f;
        H1s[(wave * 16 + mb + r) * HID + n] = (_Float16)v;
      }
    }
  }
  __syncthreads();

  // ---- stage 2: FEA = H1 @ Wout + b_out   M=128 N=64 K=256 ----
  {
    const _Float16* Arow = &H1s[(wave * 16) * HID];
    const int n0 = lane & 15, mb = (lane >> 4) * 8;
    for (int nt = 0; nt < 4; ++nt) {
      v8f c = {};
      #pragma unroll
      for (int ks = 0; ks < 8; ++ks) {
        v16h a = frag_rm(Arow + ks * 32, HID, lane);
        v16h b = frag_rm(WoutT + (nt * 16) * HID + ks * 32, HID, lane);
        c = wmma_f16(a, b, c);
      }
      const int n = nt * 16 + n0;
      const float bias = bout_s[n];
      #pragma unroll
      for (int r = 0; r < 8; ++r) {
        float v = c[r] + bias;
        _Float16 hv = (_Float16)v;
        const int m = wave * 16 + mb + r;
        Fs[m * FEA + n] = hv;
        const int e = base + m;
        if (e < M) feaH[(size_t)e * FEA + n] = hv;
      }
    }
  }
  __syncthreads();

  // ---- stage 3: gate = relu(FEA @ gW1 + gb1) @ gW2 + gb2 (fused) ----
  {
    const _Float16* Arow = &Fs[(wave * 16) * FEA];
    const int n0 = lane & 15;
    float acc[8] = {0, 0, 0, 0, 0, 0, 0, 0};
    for (int nt = 0; nt < 16; ++nt) {
      v8f c = {};
      #pragma unroll
      for (int ks = 0; ks < 2; ++ks) {
        v16h a = frag_rm(Arow + ks * 32, FEA, lane);
        v16h b = frag_rm(gW1T + (nt * 16) * FEA + ks * 32, FEA, lane);
        c = wmma_f16(a, b, c);
      }
      const int n = nt * 16 + n0;
      const float bias = gb1_s[n], w2 = gw2_s[n];
      #pragma unroll
      for (int r = 0; r < 8; ++r) {
        float v = c[r] + bias;
        v = v > 0.0f ? v : 0.0f;
        acc[r] += v * w2;
      }
    }
    // reduce the 16 lanes of each half-wave (columns of the output tile)
    #pragma unroll
    for (int r = 0; r < 8; ++r) {
      #pragma unroll
      for (int off = 1; off < 16; off <<= 1)
        acc[r] += __shfl_xor(acc[r], off, 32);
    }
    const float g2 = gb2[0];
    if ((lane & 15) == 0) {
      const int mh = (lane >> 4) * 8;
      #pragma unroll
      for (int r = 0; r < 8; ++r) {
        const int m = wave * 16 + mh + r;
        const int e = base + m;
        if (e < M) {
          const float g = acc[r] + g2;
          gateOut[e] = g;
          atomicMaxF(&gmax[sidx_s[m]], g);
        }
      }
    }
  }
}

// ---------------------------------------------------------------------------
// Kernel 3: numerators + segment-sum denominators.
// ---------------------------------------------------------------------------
__global__ void gate_norm_kernel(const float* __restrict__ gmax,
                                 const int* __restrict__ selfIdx,
                                 const int* __restrict__ nbrIdx,
                                 const float* __restrict__ atomW,
                                 float* __restrict__ gate,
                                 float* __restrict__ denom, int M) {
  int e = blockIdx.x * blockDim.x + threadIdx.x;
  if (e >= M) return;
  const int s = selfIdx[e];
  const float g = __expf(gate[e] - gmax[s]) * atomW[nbrIdx[e]];
  gate[e] = g;
  atomicAdd(&denom[s], g);
}

// ---------------------------------------------------------------------------
// Kernel 4: pooled[s,:] += (gate/denom) * fea.  self_fea_idx is SORTED, so
// each lane (owning one feature dim) accumulates a run per segment and only
// atomics at segment boundaries.
// ---------------------------------------------------------------------------
__global__ void aggregate_kernel(const float* __restrict__ gate,
                                 const float* __restrict__ denom,
                                 const int* __restrict__ selfIdx,
                                 const _Float16* __restrict__ feaH,
                                 float* __restrict__ pooled, int M) {
  const int tid = threadIdx.x;
  const int grp = tid >> 6;      // 4 groups of 64 lanes per block
  const int f   = tid & 63;      // feature dim owned by this lane
  const long eStart = ((long)blockIdx.x * 4 + grp) * 64;
  float acc = 0.0f;
  int cur = -1;
  for (int i = 0; i < 64; ++i) {
    const long e = eStart + i;
    if (e >= M) break;
    const int s = selfIdx[e];
    if (s != cur) {
      if (cur >= 0) atomicAdd(&pooled[(size_t)cur * FEA + f], acc);
      acc = 0.0f;
      cur = s;
    }
    const float coef = gate[e] / (denom[s] + 1e-13f);
    acc += coef * (float)feaH[(size_t)e * FEA + f];
  }
  if (cur >= 0) atomicAdd(&pooled[(size_t)cur * FEA + f], acc);
}

// ---------------------------------------------------------------------------
// Kernel 5: residual + relu.
// ---------------------------------------------------------------------------
__global__ void finish_kernel(const float* __restrict__ atom_fea,
                              const float* __restrict__ pooled,
                              float* __restrict__ out, int total) {
  int i = blockIdx.x * blockDim.x + threadIdx.x;
  if (i >= total) return;
  const float p = pooled[i];
  out[i] = atom_fea[i] + (p > 0.0f ? p : 0.0f);
}

// ---------------------------------------------------------------------------
extern "C" void kernel_launch(void* const* d_in, const int* in_sizes, int n_in,
                              void* d_out, int out_size, void* d_ws, size_t ws_size,
                              hipStream_t stream) {
  const float* atomW   = (const float*)d_in[0];
  const float* atomFea = (const float*)d_in[1];
  const int*   selfIdx = (const int*)d_in[2];
  const int*   nbrIdx  = (const int*)d_in[3];
  const float* W_in    = (const float*)d_in[4];
  const float* b_in    = (const float*)d_in[5];
  const float* W_out   = (const float*)d_in[6];
  const float* b_out   = (const float*)d_in[7];
  const float* gW1     = (const float*)d_in[8];
  const float* gb1     = (const float*)d_in[9];
  const float* gW2     = (const float*)d_in[10];
  const float* gb2     = (const float*)d_in[11];
  float* out = (float*)d_out;

  const int N = in_sizes[0];   // atoms
  const int M = in_sizes[2];   // edges

  // workspace layout (bytes)
  char* ws = (char*)d_ws;
  size_t off = 0;
  _Float16* WinT  = (_Float16*)(ws + off); off += (size_t)256 * 128 * 2;
  _Float16* WoutT = (_Float16*)(ws + off); off += (size_t)64 * 256 * 2;
  _Float16* gW1T  = (_Float16*)(ws + off); off += (size_t)256 * 64 * 2;
  _Float16* feaH  = (_Float16*)(ws + off); off += (size_t)M * FEA * 2;
  float* gateBuf  = (float*)(ws + off);    off += (size_t)M * 4;
  float* gmax     = (float*)(ws + off);    off += (size_t)N * 4;
  float* denom    = (float*)(ws + off);    off += (size_t)N * 4;
  float* pooled   = (float*)(ws + off);    off += (size_t)N * FEA * 4;
  (void)ws_size; (void)n_in; (void)out_size;

  convert_w_kernel<<<128, 256, 0, stream>>>(W_in, W_out, gW1, WinT, WoutT, gW1T);

  init_kernel<<<(N * FEA + 255) / 256, 256, 0, stream>>>(gmax, denom, pooled, N);

  edge_mlp_kernel<<<(M + TM - 1) / TM, 256, 0, stream>>>(
      atomFea, selfIdx, nbrIdx, WinT, WoutT, gW1T,
      b_in, b_out, gb1, gW2, gb2, feaH, gateBuf, gmax, M);

  gate_norm_kernel<<<(M + 255) / 256, 256, 0, stream>>>(
      gmax, selfIdx, nbrIdx, atomW, gateBuf, denom, M);

  aggregate_kernel<<<(M + 255) / 256, 256, 0, stream>>>(
      gateBuf, denom, selfIdx, feaH, pooled, M);

  finish_kernel<<<(N * FEA + 255) / 256, 256, 0, stream>>>(
      atomFea, pooled, out, N * FEA);
}
